// DTSemNet_54528904790526
// MI455X (gfx1250) — compile-verified, matrix operand order checked by reading, not run on hardware
//
#include <hip/hip_runtime.h>
#include <stdint.h>

// DTSemNet forward, MI455X (gfx1250, wave32).
// Stage 1: x = in_x @ W1^T + b1 via bf16 WMMA (f32 accum), 128x128 block tiles,
//          TDM (tensor_load_to_lds) double-buffered LDS staging, s_wait_tensorcnt.
// Stage 2: complete-binary-tree structure of L collapses h@L.T + segment_max to
//          z[b,leaf] = sum|x| - sum_{path} relu(+/-x) -> tree-path DP + class min.

#define IN_DIM   2048
#define N_NODES  2047
#define N_PAD    2048
#define BATCH    16384
#define OUT_DIM  10
#define HEIGHT   11
#define N_LEAVES 2048
#define KTILE    32

typedef __attribute__((ext_vector_type(16))) __bf16    v16bf;
typedef __attribute__((ext_vector_type(8)))  float     v8f;
typedef __attribute__((ext_vector_type(4)))  uint32_t  u32x4;
typedef __attribute__((ext_vector_type(8)))  int       i32x8;
typedef __attribute__((ext_vector_type(4)))  int       i32x4;

union Frag16 { uint32_t u[8]; v16bf v; };

// Packed f32 -> bf16 convert (round-to-nearest-even), one VALU op.
__device__ __forceinline__ uint32_t cvt_pk_bf16(float lo, float hi) {
  uint32_t r;
  asm("v_cvt_pk_bf16_f32 %0, %1, %2" : "=v"(r) : "v"(lo), "v"(hi));
  return r;
}

// ---------------------------------------------------------------------------
// TDM: DMA a 2D tile (tile0 x tile1 elements of 4B) from global to LDS.
// D# layout per CDNA5 ISA:
//  g0: [1:0]=count=1 | lds_addr | global_addr[56:0] | type=2
//  g1: mask=0,data_size=2(4B) | tensor_dim0/1 (OOB zero-fill) | tile_dim0/1 |
//      tensor_dim0_stride
// ---------------------------------------------------------------------------
__device__ __forceinline__ void tdm_load_2d(uint32_t lds_off, const float* gptr,
                                            uint32_t tdim0, uint32_t tdim1,
                                            uint32_t stride0,
                                            uint32_t tile0, uint32_t tile1)
{
  const uint64_t ga = (uint64_t)(uintptr_t)gptr;
  u32x4 g0;
  g0.x = 1u;                                                  // count=1 (valid)
  g0.y = lds_off;                                             // LDS byte address
  g0.z = (uint32_t)ga;                                        // global addr lo
  g0.w = ((uint32_t)(ga >> 32) & 0x1FFFFFFu) | (2u << 30);    // addr hi | type=2
  i32x8 g1;
  g1[0] = (int)(2u << 16);                                    // data_size=4B
  g1[1] = (int)((tdim0 & 0xFFFFu) << 16);                     // tensor_dim0 lo16
  g1[2] = (int)((tdim0 >> 16) | ((tdim1 & 0xFFFFu) << 16));   // dim0 hi | dim1 lo
  g1[3] = (int)((tdim1 >> 16) | (tile0 << 16));               // dim1 hi | tile_dim0
  g1[4] = (int)(tile1 & 0xFFFFu);                             // tile_dim1, tile_dim2=0
  g1[5] = (int)stride0;                                       // dim0_stride lo32
  g1[6] = 0;                                                  // stride hi, dim1_stride
  g1[7] = 0;
  const i32x4 z4 = {0, 0, 0, 0};
#if defined(__clang_major__) && (__clang_major__ >= 23)
  const i32x8 z8 = {0, 0, 0, 0, 0, 0, 0, 0};
  __builtin_amdgcn_tensor_load_to_lds(g0, g1, z4, z4, z8, 0);
#else
  __builtin_amdgcn_tensor_load_to_lds(g0, g1, z4, z4, 0);
#endif
}

// Build one 16x32 bf16 A/B fragment from an f32 LDS tile row region.
// Lane (0-15): row lr, K = k0..k0+7 (VGPR0-3) and 16+k0.. (VGPR4-7); k0=lhalf*8.
__device__ __forceinline__ Frag16 build_frag(const float* rowbase, int k0) {
  Frag16 f;
  const float4 p0 = *(const float4*)(rowbase + k0);
  const float4 p1 = *(const float4*)(rowbase + k0 + 4);
  const float4 p2 = *(const float4*)(rowbase + k0 + 16);
  const float4 p3 = *(const float4*)(rowbase + k0 + 20);
  f.u[0] = cvt_pk_bf16(p0.x, p0.y);
  f.u[1] = cvt_pk_bf16(p0.z, p0.w);
  f.u[2] = cvt_pk_bf16(p1.x, p1.y);
  f.u[3] = cvt_pk_bf16(p1.z, p1.w);
  f.u[4] = cvt_pk_bf16(p2.x, p2.y);
  f.u[5] = cvt_pk_bf16(p2.z, p2.w);
  f.u[6] = cvt_pk_bf16(p3.x, p3.y);
  f.u[7] = cvt_pk_bf16(p3.z, p3.w);
  return f;
}

// ---------------------------------------------------------------------------
// GEMM: X[m,n] = sum_k in_x[m,k] * W1[n,k] + b1[n]   (col 2047 padded)
// 256 threads = 8 waves; block tile 128x128; wave tile 32(M) x 64(N) = 2x4
// WMMA 16x16 tiles; K = 32/iter; TDM double-buffered f32 tiles in LDS.
// ---------------------------------------------------------------------------
__global__ __launch_bounds__(256) void gemm_bf16_wmma(
    const float* __restrict__ in_x, const float* __restrict__ W1,
    const float* __restrict__ b1, float* __restrict__ X)
{
  __shared__ __align__(16) float ldsA[2][128 * KTILE];  // 2 x 16KB
  __shared__ __align__(16) float ldsB[2][128 * KTILE];  // 2 x 16KB

  const int tid    = threadIdx.x;
  const int blockM = blockIdx.y * 128;
  const int blockN = blockIdx.x * 128;

  const int wave  = tid >> 5;
  const int lane  = tid & 31;
  const int lr    = lane & 15;
  const int lhalf = lane >> 4;
  const int waveM = wave >> 1;   // 0..3
  const int waveN = wave & 1;    // 0..1
  const int k0    = lhalf * 8;

  const uint32_t rowsA = (uint32_t)(BATCH - blockM);
  const uint32_t rowsB = (uint32_t)(N_NODES - blockN);  // OOB rows -> zero fill

  v8f acc[2][4];
  #pragma unroll
  for (int mt = 0; mt < 2; ++mt)
    #pragma unroll
    for (int nt = 0; nt < 4; ++nt)
      acc[mt][nt] = (v8f){0.f, 0.f, 0.f, 0.f, 0.f, 0.f, 0.f, 0.f};

  // Prologue: wave 0 DMAs the first K tile pair into buffer 0.
  if (wave == 0) {
    tdm_load_2d((uint32_t)(uintptr_t)&ldsA[0][0],
                in_x + (size_t)blockM * IN_DIM, IN_DIM, rowsA, IN_DIM, KTILE, 128);
    tdm_load_2d((uint32_t)(uintptr_t)&ldsB[0][0],
                W1 + (size_t)blockN * IN_DIM, IN_DIM, rowsB, IN_DIM, KTILE, 128);
  }

  for (int kb = 0; kb < IN_DIM / KTILE; ++kb) {
    const int buf = kb & 1;
    if (wave == 0) __builtin_amdgcn_s_wait_tensorcnt(0);
    __syncthreads();  // buf ready for all waves; buf^1 fully consumed

    if (wave == 0 && kb + 1 < IN_DIM / KTILE) {
      const int koff = (kb + 1) * KTILE;
      tdm_load_2d((uint32_t)(uintptr_t)&ldsA[buf ^ 1][0],
                  in_x + (size_t)blockM * IN_DIM + koff,
                  IN_DIM - koff, rowsA, IN_DIM, KTILE, 128);
      tdm_load_2d((uint32_t)(uintptr_t)&ldsB[buf ^ 1][0],
                  W1 + (size_t)blockN * IN_DIM + koff,
                  IN_DIM - koff, rowsB, IN_DIM, KTILE, 128);
    }

    Frag16 fa[2], fb[4];
    #pragma unroll
    for (int mt = 0; mt < 2; ++mt)
      fa[mt] = build_frag(&ldsA[buf][(waveM * 32 + mt * 16 + lr) * KTILE], k0);
    #pragma unroll
    for (int nt = 0; nt < 4; ++nt)
      fb[nt] = build_frag(&ldsB[buf][(waveN * 64 + nt * 16 + lr) * KTILE], k0);

    #pragma unroll
    for (int mt = 0; mt < 2; ++mt)
      #pragma unroll
      for (int nt = 0; nt < 4; ++nt)
        acc[mt][nt] = __builtin_amdgcn_wmma_f32_16x16x32_bf16(
            false, fa[mt].v, false, fb[nt].v, (short)0, acc[mt][nt], false, false);

    __syncthreads();  // done reading buf
  }

  // Epilogue: add bias, store f32 x. C/D layout: lane<16 -> M=r, lane>=16 -> M=8+r.
  #pragma unroll
  for (int mt = 0; mt < 2; ++mt) {
    #pragma unroll
    for (int nt = 0; nt < 4; ++nt) {
      const int m0 = blockM + waveM * 32 + mt * 16 + lhalf * 8;
      const int n0 = blockN + waveN * 64 + nt * 16 + lr;
      const float bias = (n0 < N_NODES) ? b1[n0] : 0.f;
      #pragma unroll
      for (int r = 0; r < 8; ++r)
        X[(size_t)(m0 + r) * N_PAD + n0] = acc[mt][nt][r] + bias;
    }
  }
}

// ---------------------------------------------------------------------------
// Stage 2: per batch row, S = sum|x|; each leaf walks its 11-node path:
//   pen += bit ? relu(x[node]) : relu(-x[node]);   out[b,a] = S - min_class pen
// ---------------------------------------------------------------------------
__global__ __launch_bounds__(256) void tree_dp(
    const float* __restrict__ X, const int* __restrict__ Act,
    float* __restrict__ out)
{
  __shared__ float sx[N_PAD];
  __shared__ float s_sum;
  __shared__ int   s_minbits[OUT_DIM];

  const int tid = threadIdx.x;
  const int row = blockIdx.x;
  const float* xr = X + (size_t)row * N_PAD;

  if (tid == 0) s_sum = 0.f;
  if (tid < OUT_DIM) s_minbits[tid] = 0x7F800000;  // +inf bits
  __syncthreads();

  float local = 0.f;
  for (int i = tid; i < N_PAD; i += 256) {
    const float v = (i < N_NODES) ? xr[i] : 0.f;
    sx[i] = v;
    local += fabsf(v);
  }
  #pragma unroll
  for (int off = 16; off > 0; off >>= 1)
    local += __shfl_xor(local, off, 32);
  if ((tid & 31) == 0) atomicAdd(&s_sum, local);
  __syncthreads();

  for (int leaf = tid; leaf < N_LEAVES; leaf += 256) {
    int node = 0;
    float pen = 0.f;
    #pragma unroll
    for (int lvl = HEIGHT - 1; lvl >= 0; --lvl) {
      const int bit = (leaf >> lvl) & 1;
      const float xv = sx[node];                    // near-root reads broadcast
      pen += bit ? fmaxf(xv, 0.f) : fmaxf(-xv, 0.f);
      node = 2 * node + 1 + bit;
    }
    atomicMin(&s_minbits[Act[leaf]], __float_as_int(pen));  // pen >= 0
  }
  __syncthreads();

  if (tid < OUT_DIM)
    out[(size_t)row * OUT_DIM + tid] = s_sum - __int_as_float(s_minbits[tid]);
}

// ---------------------------------------------------------------------------
extern "C" void kernel_launch(void* const* d_in, const int* in_sizes, int n_in,
                              void* d_out, int out_size, void* d_ws, size_t ws_size,
                              hipStream_t stream) {
  const float* in_x = (const float*)d_in[0];  // [16384, 2048] f32
  const float* W1   = (const float*)d_in[1];  // [2047, 2048]  f32
  const float* b1   = (const float*)d_in[2];  // [2047]        f32
  // d_in[3] (L) is realized analytically via the complete-tree structure.
  const int*   Act  = (const int*)d_in[4];    // [2048] leaf -> action

  float* X   = (float*)d_ws;                  // [16384, 2048] f32 (134 MB)
  float* out = (float*)d_out;                 // [16384, 10]   f32

  dim3 gemm_grid(N_PAD / 128, BATCH / 128);   // (16, 128)
  gemm_bf16_wmma<<<gemm_grid, 256, 0, stream>>>(in_x, W1, b1, X);
  tree_dp<<<BATCH, 256, 0, stream>>>(X, Act, out);
}
